// Int8RouterLinear_21182778703899
// MI455X (gfx1250) — matrix operand order repaired and checked by reading, not compiled, not run on hardware
//
#include <hip/hip_runtime.h>

typedef __attribute__((ext_vector_type(16))) _Float16 v16h;
typedef __attribute__((ext_vector_type(4)))  _Float16 v4h;
typedef __attribute__((ext_vector_type(8)))  float    v8f;
typedef __attribute__((ext_vector_type(4)))  float    v4f;

#define TOKENS  16384
#define HIDDEN  4096
#define EXPERTS 64

// ---------------------------------------------------------------------------
// Prepass: dequantize int8-range weights (stored as int32) -> f16 in d_ws.
// Values are integers in [-128,127]: exactly representable in f16.
// ---------------------------------------------------------------------------
__global__ __launch_bounds__(256)
void cvt_weights_i32_to_f16(const int* __restrict__ w, _Float16* __restrict__ wh) {
    int i = (blockIdx.x * blockDim.x + threadIdx.x) * 4;
    int4 v = *(const int4*)(w + i);
    v4h out = { (_Float16)(float)v.x, (_Float16)(float)v.y,
                (_Float16)(float)v.z, (_Float16)(float)v.w };
    *(v4h*)(wh + i) = out;
}

// ---------------------------------------------------------------------------
// Pipeline stage: raw f32 A chunks + four f16 B fragments (one per N subtile).
// ---------------------------------------------------------------------------
struct Stage {
    v4f  a0, a1, a2, a3;
    v16h b0, b1, b2, b3;
};

__device__ __forceinline__ void stage_load(Stage& s, const float* xk,
                                           const _Float16* wk) {
    s.a0 = *(const v4f*)(xk);
    s.a1 = *(const v4f*)(xk + 4);
    s.a2 = *(const v4f*)(xk + 8);
    s.a3 = *(const v4f*)(xk + 12);
    s.b0 = *(const v16h*)(wk);
    s.b1 = *(const v16h*)(wk + (size_t)16 * HIDDEN);
    s.b2 = *(const v16h*)(wk + (size_t)32 * HIDDEN);
    s.b3 = *(const v16h*)(wk + (size_t)48 * HIDDEN);
}

__device__ __forceinline__ void stage_mma(const Stage& s, v8f& c0, v8f& c1,
                                          v8f& c2, v8f& c3) {
    v16h a;
#pragma unroll
    for (int i = 0; i < 4; ++i) {
        a[i]      = (_Float16)s.a0[i];
        a[4 + i]  = (_Float16)s.a1[i];
        a[8 + i]  = (_Float16)s.a2[i];
        a[12 + i] = (_Float16)s.a3[i];
    }
    c0 = __builtin_amdgcn_wmma_f32_16x16x32_f16(false, a, false, s.b0, (short)0, c0, false, false);
    c1 = __builtin_amdgcn_wmma_f32_16x16x32_f16(false, a, false, s.b1, (short)0, c1, false, false);
    c2 = __builtin_amdgcn_wmma_f32_16x16x32_f16(false, a, false, s.b2, (short)0, c2, false, false);
    c3 = __builtin_amdgcn_wmma_f32_16x16x32_f16(false, a, false, s.b3, (short)0, c3, false, false);
}

// ---------------------------------------------------------------------------
// Main GEMM: out[16384,64] = x[16384,4096] @ W_f16[64,4096]^T, scaled/expert.
// One wave32 owns a 16-row x 64-expert tile; A is reused across all four
// 16-wide N subtiles so x streams from HBM exactly once. f16 weights (512 KB)
// are L2-resident. Move-free ping-pong: each stage is consumed then refilled
// in the same iteration, so loop-carried values need no register copies and
// WMMAs only ever wait on loads issued a full 64-K step earlier.
// ---------------------------------------------------------------------------
__global__ __launch_bounds__(256)
void moe_router_gemm_f16wmma(const float* __restrict__ x,
                             const _Float16* __restrict__ wh,
                             const float* __restrict__ scales,
                             float* __restrict__ out) {
    const int lane  = threadIdx.x & 31;
    const int wave  = threadIdx.x >> 5;
    const int l15   = lane & 15;
    const int half  = lane >> 4;          // 0 or 1
    const int mbase = (blockIdx.x * 8 + wave) * 16;

    // A: lane holds 16 contiguous K of row (lane&15);
    // lanes 16-31 hold K offset +16.
    const float* xrow = x + (size_t)(mbase + l15) * HIDDEN + half * 16;
    // B (contiguous K order, same as A): lane holds 16 contiguous K of column (lane&15);
    // lanes 16-31 hold K offset +16.
    const _Float16* wrow = wh + (size_t)l15 * HIDDEN + half * 16;

    v8f acc0 = {}, acc1 = {}, acc2 = {}, acc3 = {};

    Stage s0, s1;
    stage_load(s0, xrow, wrow);                    // K chunk @ 0
    stage_load(s1, xrow + 32, wrow + 32);          // K chunk @ 32

    int k = 0;
    for (; k < HIDDEN - 64; k += 64) {
        stage_mma(s0, acc0, acc1, acc2, acc3);     // consume K @ k
        stage_load(s0, xrow + k + 64, wrow + k + 64);   // refill for k+64
        stage_mma(s1, acc0, acc1, acc2, acc3);     // consume K @ k+32
        stage_load(s1, xrow + k + 96, wrow + k + 96);   // refill for k+96 (max 4064)
    }
    // tail: k == HIDDEN-64; both stages already hold the final chunks
    stage_mma(s0, acc0, acc1, acc2, acc3);
    stage_mma(s1, acc0, acc1, acc2, acc3);

    // ---- Epilogue: per-expert f32 scale; C/D layout: VGPR r -> M = r+8*half,
    // N = lane&15 ----
    const int rowb = mbase + half * 8;
    v8f accs[4] = {acc0, acc1, acc2, acc3};
#pragma unroll
    for (int nt = 0; nt < 4; ++nt) {
        const int e = nt * 16 + l15;
        const float s = scales[e];
#pragma unroll
        for (int r = 0; r < 8; ++r) {
            out[(size_t)(rowb + r) * EXPERTS + e] = accs[nt][r] * s;
        }
    }
}

extern "C" void kernel_launch(void* const* d_in, const int* in_sizes, int n_in,
                              void* d_out, int out_size, void* d_ws, size_t ws_size,
                              hipStream_t stream) {
    const float* x      = (const float*)d_in[0];
    const int*   w_i32  = (const int*)d_in[1];      // int8-range values, int32 storage
    const float* scales = (const float*)d_in[2];
    float*       out    = (float*)d_out;
    _Float16*    wh     = (_Float16*)d_ws;          // 64*4096*2 = 512 KB scratch

    // 1) Dequantize weights to f16 once (stays hot in 192 MB L2).
    {
        const int total = EXPERTS * HIDDEN;         // 262144
        const int threads = 256;
        const int blocks = total / (4 * threads);   // 256
        cvt_weights_i32_to_f16<<<blocks, threads, 0, stream>>>(w_i32, wh);
    }

    // 2) WMMA GEMM: 8 waves/block, 16 rows/wave -> 128 rows/block.
    {
        dim3 block(256);
        dim3 grid(TOKENS / 128);                    // 128 blocks, 1024 waves
        moe_router_gemm_f16wmma<<<grid, block, 0, stream>>>(x, wh, scales, out);
    }
}
